// FeatureFusionBlock_59261958750873
// MI455X (gfx1250) — 8-bit matrix operand order repaired and checked by reading, compiled, not run on hardware
//
#include <hip/hip_runtime.h>

typedef __attribute__((ext_vector_type(8))) int v8i;

#define CB 8
#define CC 256
#define CH 64
#define CW 64
#define TAPS 9
#define KCH 4
#define WSIGN_PER_CONV (TAPS * 16 * KCH * 1024)

struct PrepArgs {
    const float* w[4];
    const float* b[4];
    const float* gamma[4];
    const float* beta[4];
    const float* mean[4];
    const float* var[4];
    const float* premove[4];
};

__global__ __launch_bounds__(256) void bnn_weight_prep(PrepArgs args,
                                                       signed char* __restrict__ wsign,
                                                       float* __restrict__ alphaAll,
                                                       float* __restrict__ bconstAll) {
    const int o  = blockIdx.x;
    const int cv = blockIdx.y;
    const int ci = threadIdx.x;

    const float* w = args.w[cv];
    float wv[TAPS];
    float s9 = 0.0f;
    const int wbase = (o * CC + ci) * TAPS;
#pragma unroll
    for (int t = 0; t < TAPS; ++t) {
        wv[t] = w[wbase + t];
        s9 += fabsf(wv[t]);
    }

    __shared__ float red[256];
    red[ci] = s9;
    __syncthreads();
#pragma unroll
    for (int s = 128; s > 0; s >>= 1) {
        if (ci < s) red[ci] += red[ci + s];
        __syncthreads();
    }

    if (ci == 0) {
        const float scale = red[0] * (1.0f / 2304.0f);
        const float inv   = args.gamma[cv][o] * rsqrtf(args.var[cv][o] + 1e-5f);
        const float pm    = args.premove[cv] ? args.premove[cv][o] : 0.0f;
        alphaAll[cv * CC + o]  = scale * inv;
        bconstAll[cv * CC + o] = (args.b[cv][o] - args.mean[cv][o]) * inv +
                                 args.beta[cv][o] + pm;
    }

    const int kc   = ci >> 6;
    const int krem = ci & 63;
    const int d    = ((krem >> 5) << 2) | ((krem >> 2) & 3);
    const int hh   = (krem >> 4) & 1;
    const int bb   = krem & 3;
    const int lane = (o & 15) | (hh << 4);
    const int ot   = o >> 4;
    signed char* dst = wsign + (size_t)cv * WSIGN_PER_CONV;
#pragma unroll
    for (int t = 0; t < TAPS; ++t) {
        const float v = wv[t];
        const signed char sg = (v > 0.0f) ? 1 : ((v < 0.0f) ? -1 : 0);
        dst[((((t * 16 + ot) * KCH + kc) * 32 + lane) << 5) + d * 4 + bb] = sg;
    }
}

__global__ __launch_bounds__(256) void bnn_input_prep(const float* __restrict__ x1,
                                                      const float* __restrict__ move1,
                                                      const float* __restrict__ signmove,
                                                      float* __restrict__ U,
                                                      signed char* __restrict__ S) {
    const int idx = blockIdx.x * 256 + threadIdx.x;
    const int hw  = idx & 4095;
    const int c   = (idx >> 12) & 255;
    const int n   = idx >> 20;
    const float u = x1[idx] + move1[c];
    U[idx] = u;
    const float s = fmaxf(u, 0.0f) + signmove[c];
    const signed char sg = (s > 0.0f) ? 1 : ((s < 0.0f) ? -1 : 0);
    S[(((size_t)(n << 12) | hw) << 8) | c] = sg;
}

__global__ __launch_bounds__(256) void bnn_conv(const signed char* __restrict__ S_in,
                                                const signed char* __restrict__ Wsign,
                                                const float* __restrict__ alpha,
                                                const float* __restrict__ bconst,
                                                float* __restrict__ U,
                                                const float* __restrict__ aux,
                                                const float* __restrict__ signmove,
                                                signed char* __restrict__ S_out) {
    __shared__ int4 ldsS[1600];
    const char* ldsB = (const char*)ldsS;

    const int tid  = threadIdx.x;
    const int n    = blockIdx.z;
    const int h0   = blockIdx.y << 3;
    const int w0   = blockIdx.x << 3;

    for (int i = tid; i < 1600; i += 256) {
        const int pix = i >> 4, seg = i & 15;
        const int dy = pix / 10, dx = pix - dy * 10;
        const int hy = h0 - 1 + dy, wx = w0 - 1 + dx;
        int4 v = make_int4(0, 0, 0, 0);
        if ((unsigned)hy < (unsigned)CH && (unsigned)wx < (unsigned)CW) {
            v = *(const int4*)(S_in + ((((size_t)n * CH + hy) * CW + wx) << 8) + (seg << 4));
        }
        ldsS[i] = v;
    }
    __syncthreads();

    const int wid  = tid >> 5;
    const int lane = tid & 31;
    const int hh   = lane >> 4;
    const int ln   = lane & 15;

    int lpixBase[4];
#pragma unroll
    for (int nt = 0; nt < 4; ++nt) {
        const int p  = nt * 16 + ln;
        const int py = p >> 3, px = p & 7;
        lpixBase[nt] = py * 10 + px;
    }

    v8i acc[2][4] = {};

    const signed char* wbase0 = Wsign + ((((0 * 16 + wid * 2) * KCH) * 32 + lane) << 5);
    (void)wbase0;

    for (int t = 0; t < TAPS; ++t) {
        const int kh = t / 3, kw = t - kh * 3;
        const int tapOff = kh * 10 + kw;

        if (t + 1 < TAPS) {
            __builtin_prefetch(Wsign + (((((t + 1) * 16 + wid * 2) * KCH) * 32 + lane) << 5), 0, 1);
        }

#pragma unroll
        for (int kc = 0; kc < KCH; ++kc) {
            v8i a[2];
#pragma unroll
            for (int i = 0; i < 2; ++i) {
                const int ot = wid * 2 + i;
                a[i] = *(const v8i*)(Wsign + ((((t * 16 + ot) * KCH + kc) * 32 + lane) << 5));
            }

            int4 blo[4], bhi[4];
#pragma unroll
            for (int nt = 0; nt < 4; ++nt) {
                const int lpix = lpixBase[nt] + tapOff;
                const char* bp = ldsB + (lpix << 8) + (kc << 6) + (hh << 4);
                blo[nt] = *(const int4*)bp;
                bhi[nt] = *(const int4*)(bp + 32);
            }
#pragma unroll
            for (int nt = 0; nt < 4; ++nt) {
                v8i bv;
                bv[0] = blo[nt].x; bv[1] = blo[nt].y; bv[2] = blo[nt].z; bv[3] = blo[nt].w;
                bv[4] = bhi[nt].x; bv[5] = bhi[nt].y; bv[6] = bhi[nt].z; bv[7] = bhi[nt].w;
                acc[0][nt] = __builtin_amdgcn_wmma_i32_16x16x64_iu8(
                    true, a[0], true, bv, acc[0][nt], false, false);
                acc[1][nt] = __builtin_amdgcn_wmma_i32_16x16x64_iu8(
                    true, a[1], true, bv, acc[1][nt], false, false);
            }
        }
    }

#pragma unroll
    for (int i = 0; i < 2; ++i) {
        const int ot = wid * 2 + i;
        float al[8], bc[8], sm[8];
#pragma unroll
        for (int r = 0; r < 8; ++r) {
            const int o = ot * 16 + r + (hh << 3);
            al[r] = alpha[o];
            bc[r] = bconst[o];
            sm[r] = signmove ? signmove[o] : 0.0f;
        }
#pragma unroll
        for (int nt = 0; nt < 4; ++nt) {
            const int p  = nt * 16 + ln;
            const int py = p >> 3, px = p & 7;
            const int gh = h0 + py, gw = w0 + px;
#pragma unroll
            for (int r = 0; r < 8; ++r) {
                const int o = ot * 16 + r + (hh << 3);
                const size_t uidx = ((((size_t)n * CC + o) * CH + gh) * CW + gw);
                float u = U[uidx];
                if (aux) u += aux[uidx];
                const float outv = fmaf(al[r], (float)acc[i][nt][r], u + bc[r]);
                U[uidx] = outv;
                if (S_out) {
                    const float s = fmaxf(outv, 0.0f) + sm[r];
                    const signed char sg = (s > 0.0f) ? 1 : ((s < 0.0f) ? -1 : 0);
                    S_out[((((size_t)n * CH + gh) * CW + gw) << 8) | o] = sg;
                }
            }
        }
    }
}

__global__ __launch_bounds__(256) void bnn_upsample(const float* __restrict__ U,
                                                    float* __restrict__ out) {
    const int idx = blockIdx.x * 256 + threadIdx.x;
    const int ow = idx & 127;
    const int t  = idx >> 7;
    const int oh = t & 127;
    const int nc = t >> 7;

    float sy = oh * (63.0f / 127.0f);
    int ly = (int)sy; if (ly > 62) ly = 62;
    const float fy = sy - (float)ly;
    float sx = ow * (63.0f / 127.0f);
    int lx = (int)sx; if (lx > 62) lx = 62;
    const float fx = sx - (float)lx;

    const float* base = U + (size_t)nc * (CH * CW);
    const float v00 = base[ly * CW + lx];
    const float v01 = base[ly * CW + lx + 1];
    const float v10 = base[(ly + 1) * CW + lx];
    const float v11 = base[(ly + 1) * CW + lx + 1];
    out[idx] = (1.0f - fy) * ((1.0f - fx) * v00 + fx * v01) +
               fy * ((1.0f - fx) * v10 + fx * v11);
}

extern "C" void kernel_launch(void* const* d_in, const int* in_sizes, int n_in,
                              void* d_out, int out_size, void* d_ws, size_t ws_size,
                              hipStream_t stream) {
    (void)in_sizes; (void)n_in; (void)out_size; (void)ws_size;
    const float* x0 = (const float*)d_in[0];
    const float* x1 = (const float*)d_in[1];
    #define F(i) ((const float*)d_in[i])

    char* ws = (char*)d_ws;
    float*       U      = (float*)ws;
    signed char* Sa     = (signed char*)(ws + 33554432);
    signed char* Sb     = Sa + 8388608;
    signed char* Wsign  = Sb + 8388608;
    float*       alphaA = (float*)(Wsign + 4 * WSIGN_PER_CONV);
    float*       bconsA = alphaA + 4 * CC;

    PrepArgs pa;
    pa.w[0] = F(12); pa.b[0] = F(10); pa.beta[0] = F(2);  pa.gamma[0] = F(3);  pa.mean[0] = F(4);  pa.var[0] = F(5);  pa.premove[0] = F(17);
    pa.w[1] = F(15); pa.b[1] = F(13); pa.beta[1] = F(6);  pa.gamma[1] = F(7);  pa.mean[1] = F(8);  pa.var[1] = F(9);  pa.premove[1] = F(32);
    pa.w[2] = F(28); pa.b[2] = F(26); pa.beta[2] = F(18); pa.gamma[2] = F(19); pa.mean[2] = F(20); pa.var[2] = F(21); pa.premove[2] = F(33);
    pa.w[3] = F(31); pa.b[3] = F(29); pa.beta[3] = F(22); pa.gamma[3] = F(23); pa.mean[3] = F(24); pa.var[3] = F(25); pa.premove[3] = nullptr;

    bnn_weight_prep<<<dim3(CC, 4), 256, 0, stream>>>(pa, Wsign, alphaA, bconsA);

    bnn_input_prep<<<(CB * CC * CH * CW) / 256, 256, 0, stream>>>(x1, F(16), F(11), U, Sa);

    const dim3 cgrid(CW / 8, CH / 8, CB);
    bnn_conv<<<cgrid, 256, 0, stream>>>(Sa, Wsign + 0 * WSIGN_PER_CONV,
                                        alphaA + 0 * CC, bconsA + 0 * CC,
                                        U, nullptr, F(14), Sb);
    bnn_conv<<<cgrid, 256, 0, stream>>>(Sb, Wsign + 1 * WSIGN_PER_CONV,
                                        alphaA + 1 * CC, bconsA + 1 * CC,
                                        U, x0, F(27), Sa);
    bnn_conv<<<cgrid, 256, 0, stream>>>(Sa, Wsign + 2 * WSIGN_PER_CONV,
                                        alphaA + 2 * CC, bconsA + 2 * CC,
                                        U, nullptr, F(30), Sb);
    bnn_conv<<<cgrid, 256, 0, stream>>>(Sb, Wsign + 3 * WSIGN_PER_CONV,
                                        alphaA + 3 * CC, bconsA + 3 * CC,
                                        U, nullptr, nullptr, nullptr);

    bnn_upsample<<<(CB * CC * 128 * 128) / 256, 256, 0, stream>>>(U, (float*)d_out);
    #undef F
}
